// AlgebraicTransformerBlock_17205638988490
// MI455X (gfx1250) — compile-verified, matrix operand order checked by reading, not run on hardware
//
#include <hip/hip_runtime.h>
#include <math.h>

typedef __bf16 bf16;
typedef __attribute__((ext_vector_type(16))) __bf16 v16bf;
typedef __attribute__((ext_vector_type(8)))  float  v8f;

union FragU { uint4 u[2]; v16bf v; };

#define DMODEL 1024
#define DFFN   4096
#define NHEAD  16
#define DH     64
#define TSEQ   2048
#define BATCH  2
#define MTOT   (BATCH*TSEQ)   // 4096

// ---------------------------------------------------------------- cvt f32->bf16
__global__ void cvt_f32_bf16(const float* __restrict__ src, bf16* __restrict__ dst, int n) {
    int i = blockIdx.x * blockDim.x + threadIdx.x;
    if (i < n) dst[i] = (bf16)src[i];
}

// ---------------------------------------------------------------- algebraic LN
__device__ __forceinline__ float softplus10(float t) {
    t *= 10.0f;
    float r = (t > 20.0f) ? t : log1pf(expf(t));
    return r * 0.1f;
}

__global__ __launch_bounds__(256)
void algebraic_ln_kernel(const float* __restrict__ x,
                         const float* __restrict__ gamma,
                         const float* __restrict__ beta,
                         const float* __restrict__ a,
                         const float* __restrict__ br,
                         bf16* __restrict__ out) {
    __shared__ float ssum[8], ssq[8];
    __shared__ float stat[2]; // mean, factor
    const int row = blockIdx.x;
    const float* xr = x + (size_t)row * DMODEL;
    float vals[4];
    float s = 0.f, ss = 0.f;
#pragma unroll
    for (int k = 0; k < 4; ++k) {
        float v = xr[threadIdx.x + k * 256];
        vals[k] = v; s += v; ss += v * v;
    }
#pragma unroll
    for (int off = 16; off >= 1; off >>= 1) {
        s  += __shfl_xor(s,  off, 32);
        ss += __shfl_xor(ss, off, 32);
    }
    const int wid = threadIdx.x >> 5;
    if ((threadIdx.x & 31) == 0) { ssum[wid] = s; ssq[wid] = ss; }
    __syncthreads();
    if (threadIdx.x == 0) {
        float S = 0.f, SS = 0.f;
#pragma unroll
        for (int i = 0; i < 8; ++i) { S += ssum[i]; SS += ssq[i]; }
        const float inv = 1.0f / (float)DMODEL;
        float mean = S * inv;
        float var  = SS * inv - mean * mean;
        float z = var + 1e-5f;
        float p = a[0] + a[1] * z + a[2] * z * z;
        float q = softplus10(br[0]) + softplus10(br[1]) * z + softplus10(br[2]) * z * z;
        stat[0] = mean; stat[1] = p / q;
    }
    __syncthreads();
    const float mean = stat[0], fac = stat[1];
#pragma unroll
    for (int k = 0; k < 4; ++k) {
        int c = threadIdx.x + k * 256;
        float o = (vals[k] - mean) * fac * gamma[c] + beta[c];
        out[(size_t)row * DMODEL + c] = (bf16)o;
    }
}

// ---------------------------------------------------------------- tiled bf16 WMMA GEMM
// C[m,n] = A[m,:K] . W[n,:K] + bias[n]   (both K-contiguous)
// Explicit two-stage software pipeline (K/32 is even: 32 or 128), LDS ping-pong
// with both parities unrolled -> branchless steady state.
// MODE 0: bf16 store   1: relu -> bf16   2: bf16 store transposed as [B,H,DH,T]
// MODE 3: f32 out = resid + clip(res_scale)*val
template<int MODE>
__global__ __launch_bounds__(256)
void gemm_wmma(const bf16* __restrict__ A, const bf16* __restrict__ W,
               const float* __restrict__ bias,
               const float* __restrict__ resid,
               const float* __restrict__ res_scale,
               void* __restrict__ outp,
               int M, int N, int K) {
    constexpr int PITCH = 40;                       // 32 bf16 + 8 pad (80B rows)
    __shared__ __align__(16) bf16 As[2][128 * PITCH];
    __shared__ __align__(16) bf16 Bs[2][128 * PITCH];

    const int tid   = threadIdx.x;
    const int lane  = tid & 31;
    const int wid   = tid >> 5;
    const int wm    = wid & 3;                      // 4 waves along M
    const int wn    = wid >> 2;                     // 2 waves along N
    const int m0    = blockIdx.y * 128;
    const int n0    = blockIdx.x * 128;
    const int l15   = lane & 15;
    const int khalf = (lane >> 4) * 8;              // A/B fragment K phase
    const int mrow  = (lane >> 4) * 8;              // C fragment row phase

    // tile loader mapping: 512 x 16B segments, 2 per thread
    const int ldRow0 = tid >> 2;
    const int ldRow1 = (tid + 256) >> 2;
    const int ldSeg  = (tid & 3) * 8;

    v8f acc[2][4];
#pragma unroll
    for (int i = 0; i < 2; ++i)
#pragma unroll
        for (int j = 0; j < 4; ++j) acc[i][j] = (v8f){0,0,0,0,0,0,0,0};

    const size_t aoff0 = (size_t)(m0 + ldRow0) * K + ldSeg;
    const size_t aoff1 = (size_t)(m0 + ldRow1) * K + ldSeg;
    const size_t boff0 = (size_t)(n0 + ldRow0) * K + ldSeg;
    const size_t boff1 = (size_t)(n0 + ldRow1) * K + ldSeg;

    uint4 av0, av1, bv0, bv1;
    auto loadRegs = [&](int k1) {
        av0 = *(const uint4*)(A + aoff0 + k1);
        av1 = *(const uint4*)(A + aoff1 + k1);
        bv0 = *(const uint4*)(W + boff0 + k1);
        bv1 = *(const uint4*)(W + boff1 + k1);
    };
    auto storeTile = [&](bf16* Ad, bf16* Bd) {
        *(uint4*)(&Ad[ldRow0 * PITCH + ldSeg]) = av0;
        *(uint4*)(&Ad[ldRow1 * PITCH + ldSeg]) = av1;
        *(uint4*)(&Bd[ldRow0 * PITCH + ldSeg]) = bv0;
        *(uint4*)(&Bd[ldRow1 * PITCH + ldSeg]) = bv1;
    };
    auto computeTile = [&](const bf16* Ac, const bf16* Bc) {
        FragU af[2], bfr[4];
#pragma unroll
        for (int ms = 0; ms < 2; ++ms) {
            int r = wm * 32 + ms * 16 + l15;
            af[ms].u[0] = *(const uint4*)(Ac + r * PITCH + khalf);
            af[ms].u[1] = *(const uint4*)(Ac + r * PITCH + khalf + 16);
        }
#pragma unroll
        for (int ns = 0; ns < 4; ++ns) {
            int r = wn * 64 + ns * 16 + l15;
            bfr[ns].u[0] = *(const uint4*)(Bc + r * PITCH + khalf);
            bfr[ns].u[1] = *(const uint4*)(Bc + r * PITCH + khalf + 16);
        }
#pragma unroll
        for (int ms = 0; ms < 2; ++ms)
#pragma unroll
            for (int ns = 0; ns < 4; ++ns)
                acc[ms][ns] = __builtin_amdgcn_wmma_f32_16x16x32_bf16(
                    false, af[ms].v, false, bfr[ns].v, (short)0, acc[ms][ns], false, false);
    };

    const int ksteps = K >> 5;                      // even (32 or 128)
    loadRegs(0);
    storeTile(As[0], Bs[0]);
    __syncthreads();

    for (int ks = 0; ; ks += 2) {
        loadRegs((ks + 1) << 5);
        if (ks + 8 < ksteps) {                      // pull future rows toward L2/L0
            const int kp = (ks + 8) << 5;
            __builtin_prefetch(A + aoff0 + kp, 0, 1);
            __builtin_prefetch(W + boff0 + kp, 0, 1);
        }
        computeTile(As[0], Bs[0]);
        storeTile(As[1], Bs[1]);
        __syncthreads();
        if (ks + 2 >= ksteps) {                     // tile ks+1 is the last one
            computeTile(As[1], Bs[1]);
            break;
        }
        loadRegs((ks + 2) << 5);
        computeTile(As[1], Bs[1]);
        storeTile(As[0], Bs[0]);
        __syncthreads();
    }

    float srs = 0.f;
    if (MODE == 3) srs = fminf(fmaxf(*res_scale, 0.2f), 1.0f);

#pragma unroll
    for (int ms = 0; ms < 2; ++ms)
#pragma unroll
        for (int ns = 0; ns < 4; ++ns) {
            const int n    = n0 + wn * 64 + ns * 16 + l15;
            const float bv = bias[n];
            const int mb   = m0 + wm * 32 + ms * 16 + mrow;
#pragma unroll
            for (int r = 0; r < 8; ++r) {
                const int m = mb + r;
                float v = acc[ms][ns][r] + bv;
                if (MODE == 0) {
                    ((bf16*)outp)[(size_t)m * N + n] = (bf16)v;
                } else if (MODE == 1) {
                    ((bf16*)outp)[(size_t)m * N + n] = (bf16)fmaxf(v, 0.f);
                } else if (MODE == 2) {
                    int b = m >> 11, t = m & (TSEQ - 1);
                    int h = n >> 6,  d = n & (DH - 1);
                    ((bf16*)outp)[(((size_t)(b * NHEAD + h) * DH + d) << 11) + t] = (bf16)v;
                } else {
                    size_t idx = (size_t)m * N + n;
                    ((float*)outp)[idx] = resid[idx] + srs * v;
                }
            }
        }
}

// ---------------------------------------------------------------- relu-attention
// one wave per 16-query tile; streams keys in 32-chunks (causal-skipped)
__global__ __launch_bounds__(128)
void attn_kernel(const bf16* __restrict__ Q, const bf16* __restrict__ Kb,
                 const bf16* __restrict__ VT, const float* __restrict__ rel_emb,
                 bf16* __restrict__ Ao) {
    constexpr int WP = 40;                          // 32 + 8 pad
    __shared__ __align__(16) bf16 wbuf[4][16 * WP];

    const int lane  = threadIdx.x & 31;
    const int wid   = threadIdx.x >> 5;
    const int tile  = blockIdx.x * 4 + wid;
    const int b     = tile >> 11;                   // / (NHEAD * T/16)
    const int h     = (tile >> 7) & (NHEAD - 1);
    const int q0    = (tile & 127) * 16;
    const int l15   = lane & 15;
    const int khalf = (lane >> 4) * 8;
    const int mrow  = (lane >> 4) * 8;

    FragU qf[2];
    {
        size_t base = ((size_t)(b * TSEQ + q0 + l15) * DMODEL) + h * DH;
#pragma unroll
        for (int c = 0; c < 2; ++c) {
            qf[c].u[0] = *(const uint4*)(Q + base + c * 32 + khalf);
            qf[c].u[1] = *(const uint4*)(Q + base + c * 32 + khalf + 16);
        }
    }

    v8f accO[4];
#pragma unroll
    for (int i = 0; i < 4; ++i) accO[i] = (v8f){0,0,0,0,0,0,0,0};
    v8f rsum = (v8f){0,0,0,0,0,0,0,0};

    bf16* wl = wbuf[wid];
    const int nchunk = (q0 + 15) / 32 + 1;

    for (int ch = 0; ch < nchunk; ++ch) {
        const int j0 = ch * 32;
#pragma unroll
        for (int jt = 0; jt < 2; ++jt) {
            const int jb = j0 + jt * 16;
            if (jb > q0 + 15) {                     // fully masked subtile -> zeros
#pragma unroll
                for (int r = 0; r < 8; ++r)
                    wl[(mrow + r) * WP + jt * 16 + l15] = (bf16)0.f;
                continue;
            }
            v8f s = (v8f){0,0,0,0,0,0,0,0};
            size_t kbase = ((size_t)(b * TSEQ + jb + l15) * DMODEL) + h * DH;
#pragma unroll
            for (int c = 0; c < 2; ++c) {
                FragU kf;
                kf.u[0] = *(const uint4*)(Kb + kbase + c * 32 + khalf);
                kf.u[1] = *(const uint4*)(Kb + kbase + c * 32 + khalf + 16);
                s = __builtin_amdgcn_wmma_f32_16x16x32_bf16(
                        false, qf[c].v, false, kf.v, (short)0, s, false, false);
            }
            const int jcol = jb + l15;
#pragma unroll
            for (int r = 0; r < 8; ++r) {
                const int q   = q0 + mrow + r;
                int rel = jcol - q;
                rel = (rel < -127) ? -127 : ((rel > 127) ? 127 : rel);
                float biasv = rel_emb[(rel + 127) * NHEAD + h];
                float sv = s[r] * 0.125f + biasv;
                float w  = (jcol <= q) ? (fmaxf(sv, 0.f) + 1e-6f) : 0.f;
                rsum[r] += w;
                wl[(mrow + r) * WP + jt * 16 + l15] = (bf16)w;
            }
        }
        // re-layout w through wave-private LDS: C-layout -> A-fragment
        FragU wf;
        wf.u[0] = *(const uint4*)(wl + l15 * WP + khalf);
        wf.u[1] = *(const uint4*)(wl + l15 * WP + khalf + 16);
#pragma unroll
        for (int dt = 0; dt < 4; ++dt) {
            FragU vf;
            size_t vbase = (((size_t)(b * NHEAD + h) * DH + dt * 16 + l15) << 11) + j0;
            vf.u[0] = *(const uint4*)(VT + vbase + khalf);
            vf.u[1] = *(const uint4*)(VT + vbase + khalf + 16);
            accO[dt] = __builtin_amdgcn_wmma_f32_16x16x32_bf16(
                           false, wf.v, false, vf.v, (short)0, accO[dt], false, false);
        }
    }

#pragma unroll
    for (int r = 0; r < 8; ++r) {                   // reduce across the 16-lane half
        float v = rsum[r];
        v += __shfl_xor(v, 1, 32);
        v += __shfl_xor(v, 2, 32);
        v += __shfl_xor(v, 4, 32);
        v += __shfl_xor(v, 8, 32);
        rsum[r] = v;
    }
#pragma unroll
    for (int dt = 0; dt < 4; ++dt) {
        const int ncol = h * DH + dt * 16 + l15;
#pragma unroll
        for (int r = 0; r < 8; ++r) {
            const int q = q0 + mrow + r;
            float o = accO[dt][r] / (rsum[r] + 1e-6f);
            Ao[(size_t)(b * TSEQ + q) * DMODEL + ncol] = (bf16)o;
        }
    }
}

// ---------------------------------------------------------------- launcher
extern "C" void kernel_launch(void* const* d_in, const int* in_sizes, int n_in,
                              void* d_out, int out_size, void* d_ws, size_t ws_size,
                              hipStream_t stream) {
    (void)in_sizes; (void)n_in; (void)out_size; (void)ws_size;
    const float* x   = (const float*)d_in[0];
    const float* Wq  = (const float*)d_in[2];
    const float* bq  = (const float*)d_in[3];
    const float* Wk  = (const float*)d_in[4];
    const float* bk  = (const float*)d_in[5];
    const float* Wv  = (const float*)d_in[6];
    const float* bv  = (const float*)d_in[7];
    const float* Wo  = (const float*)d_in[8];
    const float* bo  = (const float*)d_in[9];
    const float* rel = (const float*)d_in[10];
    const float* g1  = (const float*)d_in[11];
    const float* be1 = (const float*)d_in[12];
    const float* a1  = (const float*)d_in[13];
    const float* br1 = (const float*)d_in[14];
    const float* g2  = (const float*)d_in[15];
    const float* be2 = (const float*)d_in[16];
    const float* a2  = (const float*)d_in[17];
    const float* br2 = (const float*)d_in[18];
    const float* W1  = (const float*)d_in[19];
    const float* b1  = (const float*)d_in[20];
    const float* W2  = (const float*)d_in[21];
    const float* b2  = (const float*)d_in[22];
    const float* rs  = (const float*)d_in[23];
    float* out = (float*)d_out;

    char* ws = (char*)d_ws;
    size_t off = 0;
    auto alloc = [&](size_t bytes) -> char* {
        char* p = ws + off; off += (bytes + 255) & ~(size_t)255; return p;
    };
    bf16* wq_b  = (bf16*)alloc((size_t)DMODEL * DMODEL * 2);
    bf16* wk_b  = (bf16*)alloc((size_t)DMODEL * DMODEL * 2);
    bf16* wv_b  = (bf16*)alloc((size_t)DMODEL * DMODEL * 2);
    bf16* wo_b  = (bf16*)alloc((size_t)DMODEL * DMODEL * 2);
    bf16* w1_b  = (bf16*)alloc((size_t)DFFN * DMODEL * 2);
    bf16* w2_b  = (bf16*)alloc((size_t)DMODEL * DFFN * 2);
    bf16* h1    = (bf16*)alloc((size_t)MTOT * DMODEL * 2);
    bf16* Qb    = (bf16*)alloc((size_t)MTOT * DMODEL * 2);
    bf16* Kbuf  = (bf16*)alloc((size_t)MTOT * DMODEL * 2);
    bf16* VTb   = (bf16*)alloc((size_t)MTOT * DMODEL * 2);
    bf16* Ao    = (bf16*)alloc((size_t)MTOT * DMODEL * 2);
    float* x2   = (float*)alloc((size_t)MTOT * DMODEL * 4);
    bf16* h2    = (bf16*)alloc((size_t)MTOT * DMODEL * 2);
    bf16* ffn1  = (bf16*)alloc((size_t)MTOT * DFFN * 2);

    const int nW  = DMODEL * DMODEL;   // 1M
    const int nWF = DFFN * DMODEL;     // 4M
    cvt_f32_bf16<<<(nW  + 255) / 256, 256, 0, stream>>>(Wq, wq_b, nW);
    cvt_f32_bf16<<<(nW  + 255) / 256, 256, 0, stream>>>(Wk, wk_b, nW);
    cvt_f32_bf16<<<(nW  + 255) / 256, 256, 0, stream>>>(Wv, wv_b, nW);
    cvt_f32_bf16<<<(nW  + 255) / 256, 256, 0, stream>>>(Wo, wo_b, nW);
    cvt_f32_bf16<<<(nWF + 255) / 256, 256, 0, stream>>>(W1, w1_b, nWF);
    cvt_f32_bf16<<<(nWF + 255) / 256, 256, 0, stream>>>(W2, w2_b, nWF);

    algebraic_ln_kernel<<<MTOT, 256, 0, stream>>>(x, g1, be1, a1, br1, h1);

    dim3 gP(DMODEL / 128, MTOT / 128);      // (8, 32)
    gemm_wmma<0><<<gP, 256, 0, stream>>>(h1, wq_b, bq, nullptr, nullptr, (void*)Qb,   MTOT, DMODEL, DMODEL);
    gemm_wmma<0><<<gP, 256, 0, stream>>>(h1, wk_b, bk, nullptr, nullptr, (void*)Kbuf, MTOT, DMODEL, DMODEL);
    gemm_wmma<2><<<gP, 256, 0, stream>>>(h1, wv_b, bv, nullptr, nullptr, (void*)VTb,  MTOT, DMODEL, DMODEL);

    attn_kernel<<<(BATCH * NHEAD * (TSEQ / 16)) / 4, 128, 0, stream>>>(Qb, Kbuf, VTb, rel, Ao);

    gemm_wmma<3><<<gP, 256, 0, stream>>>(Ao, wo_b, bo, x, rs, (void*)x2, MTOT, DMODEL, DMODEL);

    algebraic_ln_kernel<<<MTOT, 256, 0, stream>>>(x2, g2, be2, a2, br2, h2);

    dim3 gF1(DFFN / 128, MTOT / 128);       // (32, 32)
    gemm_wmma<1><<<gF1, 256, 0, stream>>>(h2, w1_b, b1, nullptr, nullptr, (void*)ffn1, MTOT, DFFN, DMODEL);
    gemm_wmma<3><<<gP, 256, 0, stream>>>(ffn1, w2_b, b2, x2, rs, (void*)out, MTOT, DMODEL, DFFN);
}
